// VisualRelationEncoder_47347719471340
// MI455X (gfx1250) — matrix operand order repaired and checked by reading, not compiled, not run on hardware
//
#include <hip/hip_runtime.h>
#include <hip/hip_bf16.h>

// ---------------- types / constants ----------------
typedef __attribute__((ext_vector_type(16))) __bf16 v16bf;
typedef __attribute__((ext_vector_type(8)))  float  v8f;
typedef int v4i_vs __attribute__((__vector_size__(4 * sizeof(int))));

#define EPSV 1e-5f
static constexpr int    kCH = 640;          // NI*NO channels for BatchNorm1d
static constexpr int    kR  = 20480;        // B*NI*NO rows
static constexpr long   kRH = (long)kR * 512;

#if defined(__has_builtin)
#if __has_builtin(__builtin_amdgcn_global_load_async_to_lds_b128)
#define HAVE_ASYNC_LDS 1
#endif
#endif

#define ASG __attribute__((address_space(1)))
#define ASL __attribute__((address_space(3)))

// ---------------- helpers ----------------
__device__ __forceinline__ float wsum(float v) {
#pragma unroll
  for (int m = 16; m > 0; m >>= 1) v += __shfl_xor(v, m, 32);
  return v;
}

__device__ __forceinline__ unsigned int f2bf_pack(float x0, float x1) {
  unsigned int u0 = __float_as_uint(x0), u1 = __float_as_uint(x1);
  u0 = (u0 + 0x7FFFu + ((u0 >> 16) & 1u)) >> 16;
  u1 = (u1 + 0x7FFFu + ((u1 >> 16) & 1u)) >> 16;
  return (u1 << 16) | (u0 & 0xFFFFu);
}

#ifdef HAVE_ASYNC_LDS
__device__ __forceinline__ void async_copy16(const void* g, void* l) {
  // flat->AS1: identical bits; flat->AS3: LDS offset lives in addr[31:0] (ISA 10.2)
  ASG v4i_vs* gp = (ASG v4i_vs*)(unsigned long long)g;
  ASL v4i_vs* lp = (ASL v4i_vs*)(unsigned int)(unsigned long long)l;
  __builtin_amdgcn_global_load_async_to_lds_b128(gp, lp, 0, 0);
}
__device__ __forceinline__ void wait_async0() {
#if __has_builtin(__builtin_amdgcn_s_wait_asynccnt)
  __builtin_amdgcn_s_wait_asynccnt(0);
#else
  asm volatile("s_wait_asynccnt 0x0" ::: "memory");
#endif
}
#endif

// ---------------- BatchNorm stats: per-channel over [32, 640, L] ----------------
// Emits folded affine: s2[c] = g[c]*rsqrt(var+eps), t2[c] = b[c] - s2[c]*mean
__global__ void k_bnstats(const float* __restrict__ X, int L,
                          const float* __restrict__ g, const float* __restrict__ b,
                          float* __restrict__ s2, float* __restrict__ t2) {
  int c = blockIdx.x, tid = threadIdx.x;
  int n = 32 * L;
  float s = 0.f, ss = 0.f;
  for (int i = tid; i < n; i += blockDim.x) {
    int bb = i / L, l = i - bb * L;
    float x = X[((size_t)bb * kCH + c) * L + l];
    s += x; ss += x * x;
  }
  __shared__ float r1[8], r2[8];
  s = wsum(s); ss = wsum(ss);
  int lane = tid & 31, w = tid >> 5;
  if (lane == 0) { r1[w] = s; r2[w] = ss; }
  __syncthreads();
  if (tid == 0) {
    float S = 0.f, SS = 0.f;
    int nw = blockDim.x >> 5;
    for (int i = 0; i < nw; i++) { S += r1[i]; SS += r2[i]; }
    float m = S / n, var = SS / n - m * m;
    float sc = g[c] * rsqrtf(var + EPSV);
    s2[c] = sc; t2[c] = b[c] - sc * m;
  }
}

// apply y = relu(s2[c]*x + t2[c]) in place on [R, W]
__global__ void k_bn_relu(float* __restrict__ X, int W,
                          const float* __restrict__ s2, const float* __restrict__ t2,
                          long n) {
  long i = (long)blockIdx.x * blockDim.x + threadIdx.x;
  if (i >= n) return;
  int c = (int)((i / W) % kCH);
  X[i] = fmaxf(s2[c] * X[i] + t2[c], 0.f);
}

// ---------------- per-row LayerNorm stats over (optionally split) row ----------------
__global__ void k_rowstats(const float* __restrict__ A0, int ld0, int d0,
                           const float* __restrict__ A1, int ld1, int d1,
                           float* __restrict__ rm, float* __restrict__ rr) {
  int r = blockIdx.x, tid = threadIdx.x;
  float s = 0.f, ss = 0.f;
  for (int j = tid; j < d0; j += blockDim.x) { float x = A0[(size_t)r * ld0 + j]; s += x; ss += x * x; }
  if (A1) for (int j = tid; j < d1; j += blockDim.x) { float x = A1[(size_t)r * ld1 + j]; s += x; ss += x * x; }
  __shared__ float r1[4], r2[4];
  s = wsum(s); ss = wsum(ss);
  int lane = tid & 31, w = tid >> 5;
  if (lane == 0) { r1[w] = s; r2[w] = ss; }
  __syncthreads();
  if (tid == 0) {
    float S = 0.f, SS = 0.f;
    for (int i = 0; i < (int)(blockDim.x >> 5); i++) { S += r1[i]; SS += r2[i]; }
    int D = d0 + d1;
    float m = S / D, var = SS / D - m * m;
    rm[r] = m; rr[r] = rsqrtf(var + EPSV);
  }
}

// x = relu((x-rm[r])*rr[r]*g[j] + b[j]) in place on [R, D]
__global__ void k_ln_relu(float* __restrict__ X, int D,
                          const float* __restrict__ g, const float* __restrict__ b,
                          const float* __restrict__ rm, const float* __restrict__ rr,
                          long n) {
  long i = (long)blockIdx.x * blockDim.x + threadIdx.x;
  if (i >= n) return;
  int r = (int)(i / D), j = (int)(i - (long)r * D);
  X[i] = fmaxf((X[i] - rm[r]) * rr[r] * g[j] + b[j], 0.f);
}

// ---------------- weight prep ----------------
__global__ void k_colsum(const float* __restrict__ W, int K, int N, float* __restrict__ cs) {
  int j = blockIdx.x * blockDim.x + threadIdx.x;
  if (j >= N) return;
  float s = 0.f;
  for (int k = 0; k < K; k++) s += W[(size_t)k * N + j];
  cs[j] = s;
}

__global__ void k_w2bf(const float* __restrict__ W, int n, unsigned short* __restrict__ O) {
  int i = blockIdx.x * blockDim.x + threadIdx.x;
  if (i >= n) return;
  unsigned int u = __float_as_uint(W[i]);
  O[i] = (unsigned short)((u + 0x7FFFu + ((u >> 16) & 1u)) >> 16);
}

// ---------------- WMMA GEMM: C[M,N] = epilogue(Atransform[M,K] @ Wbf16[K,N]) ----------
// Block tile 128(M) x 64(N), 4 waves; each wave: 32 rows x 64 cols = 8 WMMA / K-step.
// mode 0: A element = A0[r*ld0+k]
// mode 1: A element = ((k<split? A0[r*ld0+k] : A1[r*ld1+k-split]) - rm[r])*rr[r]*lg[k]+lb2[k]
// ep 0: C = acc + bias[j];  ep 1: C = relu(acc + bias[j]);
// ep 2 (folded BN1): C = chs[r%640]*acc + cht[r%640]*cs[j] + bias[j]
__global__ __launch_bounds__(128) void k_gemm(
    const float* __restrict__ A0, int ld0,
    const float* __restrict__ A1, int ld1, int split,
    const float* __restrict__ rm, const float* __restrict__ rr,
    const float* __restrict__ lg, const float* __restrict__ lb2,
    const unsigned short* __restrict__ Wb, int K, int N,
    const float* __restrict__ bias, const float* __restrict__ cs,
    const float* __restrict__ chs, const float* __restrict__ cht,
    int mode, int ep, float* __restrict__ C) {
  __shared__ alignas(16) unsigned int As[128 * 16];  // 128 rows x 32 K (bf16 pairs)
  __shared__ alignas(16) unsigned int Bs[32 * 32];   // 32 K x 64 N (bf16 pairs)
  const int tid = threadIdx.x, lane = tid & 31, wv = tid >> 5;
  const int tN = blockIdx.x * 64, tM = blockIdx.y * 128;
  union UC { v8f v; float f[8]; } acc[2][4];
#pragma unroll
  for (int h = 0; h < 2; h++)
#pragma unroll
    for (int s = 0; s < 4; s++)
#pragma unroll
      for (int i = 0; i < 8; i++) acc[h][s].f[i] = 0.f;

  for (int kk = 0; kk < K; kk += 32) {
    // ---- B tile: 32x64 bf16 = 256 x 16B chunks (pure copy -> async DMA to LDS)
#pragma unroll
    for (int i = 0; i < 2; i++) {
      int cidx = i * 128 + tid;
      int rB = cidx >> 3, c8 = cidx & 7;  // 8 chunks of 16B per 128B row
      const void* g = (const char*)Wb + (((size_t)(kk + rB) * N + tN) << 1) + c8 * 16;
      void* l = (void*)&Bs[rB * 32 + c8 * 4];
#ifdef HAVE_ASYNC_LDS
      async_copy16(g, l);
#else
      *(uint4*)l = *(const uint4*)g;
#endif
    }
    // ---- A tile: 128x32, fp32 load + transform + RNE bf16 pack (overlaps async B)
#pragma unroll
    for (int i = 0; i < 16; i++) {
      int u = i * 128 + tid;
      int row = u >> 4, w = u & 15;
      int gr = tM + row, k0 = kk + 2 * w;
      float x0, x1;
      if (mode == 0) {
        const float* p = &A0[(size_t)gr * ld0 + k0];
        x0 = p[0]; x1 = p[1];
      } else {
        float m = rm[gr], rs = rr[gr];
        int ka = k0, kb = k0 + 1;
        float y0 = (ka < split) ? A0[(size_t)gr * ld0 + ka] : A1[(size_t)gr * ld1 + (ka - split)];
        float y1 = (kb < split) ? A0[(size_t)gr * ld0 + kb] : A1[(size_t)gr * ld1 + (kb - split)];
        x0 = (y0 - m) * rs * lg[ka] + lb2[ka];
        x1 = (y1 - m) * rs * lg[kb] + lb2[kb];
      }
      As[u] = f2bf_pack(x0, x1);
    }
#ifdef HAVE_ASYNC_LDS
    wait_async0();
#endif
    __syncthreads();

    union UF { v16bf v; unsigned int u[8]; } fa0, fa1, fb;
    const int hs = lane >> 4, lr = lane & 15;
#pragma unroll
    for (int d = 0; d < 8; d++) {          // A 16x32 bf16 lane layout
      int w = (d >> 2) * 8 + hs * 4 + (d & 3);
      fa0.u[d] = As[(wv * 32 + lr) * 16 + w];
      fa1.u[d] = As[(wv * 32 + 16 + lr) * 16 + w];
    }
#pragma unroll
    for (int s = 0; s < 4; s++) {
#pragma unroll
      for (int d = 0; d < 8; d++) {        // B 32x16: lane = N column, dwords = K pairs (table order)
        int kb = (d >> 2) * 16 + hs * 8 + 2 * (d & 3);
        const unsigned short* Bh = (const unsigned short*)Bs;
        fb.u[d] = (unsigned int)Bh[kb * 64 + s * 16 + lr] | ((unsigned int)Bh[(kb + 1) * 64 + s * 16 + lr] << 16);
      }
      acc[0][s].v = __builtin_amdgcn_wmma_f32_16x16x32_bf16(
          false, fa0.v, false, fb.v, (short)0, acc[0][s].v, false, false);
      acc[1][s].v = __builtin_amdgcn_wmma_f32_16x16x32_bf16(
          false, fa1.v, false, fb.v, (short)0, acc[1][s].v, false, false);
    }
    __syncthreads();
  }

  const int colBase = tN + (lane & 15);
#pragma unroll
  for (int h = 0; h < 2; h++) {
    int rowBase = tM + wv * 32 + h * 16 + (lane >> 4) * 8;
#pragma unroll
    for (int s = 0; s < 4; s++) {
      int col = colBase + s * 16;
      float bv = bias ? bias[col] : 0.f;
      float csv = (ep == 2) ? cs[col] : 0.f;
#pragma unroll
      for (int i = 0; i < 8; i++) {
        int row = rowBase + i;
        float v = acc[h][s].f[i];
        if (ep == 2) { int c = row % kCH; v = chs[c] * v + cht[c] * csv + bv; }
        else { v += bv; if (ep == 1) v = fmaxf(v, 0.f); }
        C[(size_t)row * N + col] = v;
      }
    }
  }
}

// ---------------- spatial linear (K=6): out = BN1fold(x) @ W6 + b ----------------
__global__ void k_splin(const float* __restrict__ X,
                        const float* __restrict__ s2, const float* __restrict__ t2,
                        const float* __restrict__ W6, const float* __restrict__ bb,
                        float* __restrict__ O) {
  int i = blockIdx.x * blockDim.x + threadIdx.x;
  if (i >= kR * 64) return;
  int r = i >> 6, j = i & 63;
  int c = r % kCH;
  float s = s2[c], t = t2[c];
  float a = bb[j];
#pragma unroll
  for (int k = 0; k < 6; k++) a += (s * X[(size_t)r * 6 + k] + t) * W6[k * 64 + j];
  O[i] = a;
}

// ---------------- batched 10x10 scores + row softmax: S = softmax(scale * Q K^T) ---
__global__ __launch_bounds__(128) void k_scores(const float* __restrict__ Q,
                                                const float* __restrict__ Kk,
                                                float scale, float* __restrict__ S) {
  int b = blockIdx.x, tid = threadIdx.x;
  __shared__ float sq[10 * 512];
  __shared__ float sk[10 * 512];
  __shared__ float sraw[100];
  const float* qb = &Q[(size_t)b * 10 * 512];
  const float* kb = &Kk[(size_t)b * 10 * 512];
  for (int i = tid; i < 5120; i += 128) { sq[i] = qb[i]; sk[i] = kb[i]; }
  __syncthreads();
  int lane = tid & 31, wv = tid >> 5;
  for (int p = wv; p < 100; p += 4) {
    int n = p / 10, m = p - n * 10;
    float a = 0.f;
    for (int d = lane; d < 512; d += 32) a += sq[n * 512 + d] * sk[m * 512 + d];
    a = wsum(a);
    if (lane == 0) sraw[p] = a * scale;
  }
  __syncthreads();
  if (tid < 10) {
    int n = tid;
    float mx = -1e30f;
    for (int m = 0; m < 10; m++) mx = fmaxf(mx, sraw[n * 10 + m]);
    float e[10], sm = 0.f;
    for (int m = 0; m < 10; m++) { e[m] = __expf(sraw[n * 10 + m] - mx); sm += e[m]; }
    float inv = 1.f / sm;
    for (int m = 0; m < 10; m++) S[(size_t)b * 100 + n * 10 + m] = e[m] * inv;
  }
}

// ---------------- batched aggregation: O[b] = (relu?)(A[b](10x10) @ X[b](10x512) + bias)
__global__ void k_bav(const float* __restrict__ A, const float* __restrict__ X,
                      const float* __restrict__ bias, int doRelu, float* __restrict__ O) {
  int b = blockIdx.x, tid = threadIdx.x;
  __shared__ float sA[100];
  if (tid < 100) sA[tid] = A[(size_t)b * 100 + tid];
  __syncthreads();
  for (int j = tid; j < 512; j += blockDim.x) {
    float xm[10];
#pragma unroll
    for (int m = 0; m < 10; m++) xm[m] = X[((size_t)b * 10 + m) * 512 + j];
    float bv = bias ? bias[j] : 0.f;
#pragma unroll
    for (int n = 0; n < 10; n++) {
      float a = bv;
#pragma unroll
      for (int m = 0; m < 10; m++) a += sA[n * 10 + m] * xm[m];
      if (doRelu) a = fmaxf(a, 0.f);
      O[((size_t)b * 10 + n) * 512 + j] = a;
    }
  }
}

// ---------------- relation bias + zero-degree mask + renormalizing softmax ---------
__global__ void k_attn2(float* __restrict__ S, const float* __restrict__ adj,
                        const float* __restrict__ bw, const float* __restrict__ bb) {
  int t = blockIdx.x * blockDim.x + threadIdx.x;
  if (t >= kR) return;
  float b0 = bb[0];
  float sc[10];
  for (int m = 0; m < 10; m++) {
    const float* p = &adj[((size_t)t * 10 + m) * 11];
    float deg = 0.f, rb = 0.f;
#pragma unroll
    for (int q = 0; q < 11; q++) { deg += p[q]; rb += p[q] * bw[q]; }
    rb += b0;
    float a = S[(size_t)t * 10 + m];
    sc[m] = ((deg == 0.f) ? -1e10f : a) + rb;
  }
  float mx = -1e30f;
  for (int m = 0; m < 10; m++) mx = fmaxf(mx, sc[m]);
  float e[10], sm = 0.f;
  for (int m = 0; m < 10; m++) { e[m] = __expf(sc[m] - mx); sm += e[m]; }
  float inv = 1.f / sm;
  for (int m = 0; m < 10; m++) S[(size_t)t * 10 + m] = e[m] * inv;
}

// ---------------- final blend: out = softmax(factor)[0]*node_a + [1]*node ----------
__global__ void k_final(float* __restrict__ out, const float* __restrict__ node,
                        const float* __restrict__ factor, long n) {
  long i = (long)blockIdx.x * blockDim.x + threadIdx.x;
  if (i >= n) return;
  float f0 = factor[0], f1 = factor[1];
  float m = fmaxf(f0, f1);
  float e0 = __expf(f0 - m), e1 = __expf(f1 - m);
  float inv = 1.f / (e0 + e1);
  out[i] = (e0 * inv) * out[i] + (e1 * inv) * node[i];
}

// ====================================================================================
extern "C" void kernel_launch(void* const* d_in, const int* in_sizes, int n_in,
                              void* d_out, int out_size, void* d_ws, size_t ws_size,
                              hipStream_t stream) {
  (void)in_sizes; (void)n_in; (void)out_size; (void)ws_size;
#define F32(i) ((const float*)d_in[i])
  const float* obj  = F32(0);
  const float* attr = F32(1);
  const float* spf  = F32(2);
  const float* adjm = F32(3);

  // workspace layout (floats)
  float* ws = (float*)d_ws;
  size_t RS = (size_t)kR * 512;
  float* F0 = ws;               // obj activation
  float* F1 = F0 + RS;          // attr / v / Xs
  float* F2 = F1 + RS;          // a / k / ctx / node
  float* F3 = F2 + RS;          // h / o
  float* F4 = F3 + RS;          // Xa / q / o2
  float* SP = F4 + RS;          // [R,64] spatial feats
  float* P0 = SP + (size_t)kR * 64;   // [2048,100] adj/attn
  float* rm = P0 + 2048 * 100;
  float* rr = rm + kR;
  float* chs = rr + kR;
  float* cht = chs + kCH;
  float* csv = cht + kCH;       // 512 colsums
  unsigned short* Wbuf = (unsigned short*)(csv + 512);  // 2MB bf16 weight staging (16B aligned)

  const dim3 gEW((unsigned)((kRH + 255) / 256));
  auto cvt = [&](const float* W, int n) {
    k_w2bf<<<(n + 255) / 256, 256, 0, stream>>>(W, n, Wbuf);
  };
  auto gemm = [&](const float* A0, int ld0, const float* A1, int ld1, int split,
                  const float* rm_, const float* rr_, const float* lg, const float* lb2,
                  int K, const float* bias, int mode, int ep, float* C) {
    k_gemm<<<dim3(512 / 64, kR / 128), 128, 0, stream>>>(
        A0, ld0, A1, ld1, split, rm_, rr_, lg, lb2, Wbuf, K, 512,
        bias, csv, chs, cht, mode, ep, C);
  };

  // ---- obj branch: BN1 folded into GEMM, then BN2+ReLU
  k_bnstats<<<kCH, 256, 0, stream>>>(obj, 2048, F32(4), F32(5), chs, cht);
  k_colsum<<<2, 256, 0, stream>>>(F32(6), 2048, 512, csv);
  cvt(F32(6), 2048 * 512);
  gemm(obj, 2048, nullptr, 0, 1 << 30, nullptr, nullptr, nullptr, nullptr,
       2048, F32(7), 0, 2, F0);
  k_bnstats<<<kCH, 256, 0, stream>>>(F0, 512, F32(8), F32(9), chs, cht);
  k_bn_relu<<<gEW, 256, 0, stream>>>(F0, 512, chs, cht, kRH);

  // ---- attr branch
  k_bnstats<<<kCH, 256, 0, stream>>>(attr, 768, F32(10), F32(11), chs, cht);
  k_colsum<<<2, 256, 0, stream>>>(F32(12), 768, 512, csv);
  cvt(F32(12), 768 * 512);
  gemm(attr, 768, nullptr, 0, 1 << 30, nullptr, nullptr, nullptr, nullptr,
       768, F32(13), 0, 2, F1);
  k_bnstats<<<kCH, 256, 0, stream>>>(F1, 512, F32(14), F32(15), chs, cht);
  k_bn_relu<<<gEW, 256, 0, stream>>>(F1, 512, chs, cht, kRH);

  // ---- combine1: LN(concat[obj,attr]) @ c1_W, then LN2+ReLU -> a (F2)
  k_rowstats<<<kR, 128, 0, stream>>>(F0, 512, 512, F1, 512, 512, rm, rr);
  cvt(F32(24), 1024 * 512);
  gemm(F0, 512, F1, 512, 512, rm, rr, F32(22), F32(23), 1024, F32(25), 1, 0, F2);
  k_rowstats<<<kR, 128, 0, stream>>>(F2, 512, 512, nullptr, 0, 0, rm, rr);
  k_ln_relu<<<gEW, 256, 0, stream>>>(F2, 512, F32(26), F32(27), rm, rr, kRH);

  // ---- edge learner + GCN-a: node_a -> d_out
  cvt(F32(34), 512 * 512);
  gemm(F2, 512, nullptr, 0, 1 << 30, nullptr, nullptr, nullptr, nullptr,
       512, F32(35), 0, 1, F3);                                  // h = relu(a@al_W+b)
  cvt(F32(36), 512 * 512);
  gemm(F2, 512, nullptr, 0, 1 << 30, nullptr, nullptr, nullptr, nullptr,
       512, nullptr, 0, 0, F4);                                  // Xa = a@gcn_a_W
  k_scores<<<2048, 128, 0, stream>>>(F3, F3, 1.0f, P0);          // adj_a
  k_bav<<<2048, 256, 0, stream>>>(P0, F4, F32(37), 1, (float*)d_out); // node_a

  // ---- spatial branch -> SP [R,64]
  k_bnstats<<<kCH, 256, 0, stream>>>(spf, 6, F32(16), F32(17), chs, cht);
  k_splin<<<(kR * 64 + 255) / 256, 256, 0, stream>>>(spf, chs, cht, F32(18), F32(19), SP);
  k_bnstats<<<kCH, 256, 0, stream>>>(SP, 64, F32(20), F32(21), chs, cht);
  k_bn_relu<<<(kR * 64 + 255) / 256, 256, 0, stream>>>(SP, 64, chs, cht, (long)kR * 64);

  // ---- combine2: LN(concat[obj,SP]) @ c2_W, LN2+ReLU -> o (F3)
  k_rowstats<<<kR, 128, 0, stream>>>(F0, 512, 512, SP, 64, 64, rm, rr);
  cvt(F32(30), 576 * 512);
  gemm(F0, 512, SP, 64, 512, rm, rr, F32(28), F32(29), 576, F32(31), 1, 0, F3);
  k_rowstats<<<kR, 128, 0, stream>>>(F3, 512, 512, nullptr, 0, 0, rm, rr);
  k_ln_relu<<<gEW, 256, 0, stream>>>(F3, 512, F32(32), F32(33), rm, rr, kRH);

  // ---- single-head self-attention
  cvt(F32(40), 512 * 512);
  gemm(F3, 512, nullptr, 0, 1 << 30, nullptr, nullptr, nullptr, nullptr, 512, F32(41), 0, 0, F4); // q
  cvt(F32(42), 512 * 512);
  gemm(F3, 512, nullptr, 0, 1 << 30, nullptr, nullptr, nullptr, nullptr, 512, F32(43), 0, 0, F2); // k
  cvt(F32(44), 512 * 512);
  gemm(F3, 512, nullptr, 0, 1 << 30, nullptr, nullptr, nullptr, nullptr, 512, F32(45), 0, 0, F1); // v
  k_scores<<<2048, 128, 0, stream>>>(F4, F2, 0.04419417382415922f, P0);  // attn
  k_bav<<<2048, 256, 0, stream>>>(P0, F1, nullptr, 0, F2);               // ctx = attn@v
  cvt(F32(46), 512 * 512);
  gemm(F2, 512, nullptr, 0, 1 << 30, nullptr, nullptr, nullptr, nullptr, 512, F32(47), 0, 0, F4); // o2

  // ---- GCN-s with relation-bias softmax
  cvt(F32(38), 512 * 512);
  gemm(F4, 512, nullptr, 0, 1 << 30, nullptr, nullptr, nullptr, nullptr, 512, nullptr, 0, 0, F1); // Xs
  k_attn2<<<(kR + 127) / 128, 128, 0, stream>>>(P0, adjm, F32(48), F32(49));  // adj2
  k_bav<<<2048, 256, 0, stream>>>(P0, F1, F32(39), 1, F2);                    // node

  // ---- final blend
  k_final<<<gEW, 256, 0, stream>>>((float*)d_out, F2, F32(50), kRH);
#undef F32
}
